// MultiHeadedAttention_72877005078647
// MI455X (gfx1250) — matrix operand order repaired and checked by reading, not compiled, not run on hardware
//
#include <hip/hip_runtime.h>
#include <hip/hip_bf16.h>

// ---------------------------------------------------------------- constants
constexpr int cB  = 2;
constexpr int cS  = 1024;
constexpr int cD  = 1024;
constexpr int cH  = 16;
constexpr int cDH = 64;
constexpr int cT  = 65;          // 2*MAX_REL+1 relative buckets
constexpr int cBH = cB * cH;     // 32
constexpr int cBS = cB * cS;     // 2048 rows in projection GEMMs

typedef __attribute__((ext_vector_type(16))) __bf16 v16bf;
typedef __attribute__((ext_vector_type(8)))  __bf16 v8bf;
typedef __attribute__((ext_vector_type(8)))  float  v8f;

// ------------------------------------------------ CDNA5 async global -> LDS
// GLOBAL_LOAD_ASYNC_TO_LDS_B128 (ASYNCcnt-tracked). VDST = VGPR holding LDS
// byte offset, VADDR = 64-bit global address. Portable inline asm (bypasses
// the toolchain-divergent TDM builtins).
__device__ __forceinline__ void async_ld_lds_b128(void* lds_dst, const void* gsrc) {
    uint32_t loff =
        (uint32_t)(size_t)(__attribute__((address_space(3))) char*)lds_dst;
    asm volatile("global_load_async_to_lds_b128 %0, %1, off"
                 :: "v"(loff), "v"((unsigned long long)(size_t)gsrc)
                 : "memory");
}
__device__ __forceinline__ void wait_asynccnt0() {
    asm volatile("s_wait_asynccnt 0x0" ::: "memory");
}

// -------------------------------------------------- WMMA fragment loaders
// A-fragment 16x32 bf16 (M x K), contiguous k order (B loaders use the same):
//   lane<16 : M=lane,    K = k0..k0+15
//   lane>=16: M=lane-16, K = k0+16..k0+31
__device__ __forceinline__ v16bf load_a_bf16(const __bf16* __restrict__ A,
                                             int lda, int m0, int k0, int lane) {
    int row = m0 + (lane & 15);
    int kh  = (lane >> 4) * 16;
    const __bf16* p = A + (size_t)row * lda + k0 + kh;
    v8bf lo = *(const v8bf*)p;          // 16B contiguous
    v8bf hi = *(const v8bf*)(p + 8);    // 16B contiguous
    v16bf f;
#pragma unroll
    for (int i = 0; i < 8; ++i) { f[i] = lo[i]; f[i + 8] = hi[i]; }
    return f;
}

// Same A-fragment but converting from fp32 source on the fly.
__device__ __forceinline__ v16bf load_a_f32(const float* __restrict__ A,
                                            int lda, int m0, int k0, int lane) {
    int row = m0 + (lane & 15);
    int kh  = (lane >> 4) * 16;
    const float* p = A + (size_t)row * lda + k0 + kh;
    v16bf f;
#pragma unroll
    for (int i = 0; i < 8; ++i) { f[i] = (__bf16)p[i]; f[i + 8] = (__bf16)p[i + 8]; }
    return f;
}

// B-fragment 32x16 bf16 (K x N) from a [N,K]-stored (transposed) matrix:
//   lane<16 : N=lane,    K = k0..k0+15   (32B contiguous)
//   lane>=16: N=lane-16, K = k0+16..k0+31
__device__ __forceinline__ v16bf load_b_bf16T(const __bf16* __restrict__ BT,
                                              int ldb, int n0, int k0, int lane) {
    int n  = n0 + (lane & 15);
    int kb = k0 + ((lane >> 4) * 16);
    return *(const v16bf*)(BT + (size_t)n * ldb + kb);
}

// B-fragment from an LDS-staged [64][32] bf16 tile (n-local major).
__device__ __forceinline__ v16bf load_b_lds(const __bf16* tile, int nloc, int lane) {
    int n  = nloc + (lane & 15);
    int kb = (lane >> 4) * 16;
    return *(const v16bf*)(tile + n * 32 + kb);
}

#define WMMA_BF16(a, b, c) \
    __builtin_amdgcn_wmma_f32_16x16x32_bf16(false, (a), false, (b), (short)0, (c), false, false)

// ---------------------------------------------------------------- casts
__global__ void mha_castx_kernel(const float* __restrict__ X, __bf16* __restrict__ XB, int n) {
    int i = blockIdx.x * blockDim.x + threadIdx.x;
    if (i < n) XB[i] = (__bf16)X[i];
}

// LDS-tiled transpose+cast: WT[n][k] = (bf16)W[k][n], W is 1024x1024.
__global__ __launch_bounds__(256) void mha_castT_kernel(const float* __restrict__ W,
                                                        __bf16* __restrict__ WT) {
    __shared__ float tile[32][33];
    int tx = threadIdx.x & 31;
    int ty = threadIdx.x >> 5;          // 0..7
    int k0 = blockIdx.x * 32;
    int n0 = blockIdx.y * 32;
#pragma unroll
    for (int r = 0; r < 32; r += 8)
        tile[ty + r][tx] = W[(size_t)(k0 + ty + r) * cD + n0 + tx];
    __syncthreads();
#pragma unroll
    for (int r = 0; r < 32; r += 8)
        WT[(size_t)(n0 + ty + r) * cD + k0 + tx] = (__bf16)tile[tx][ty + r];
}

// ------------------------------------------------------- generic WMMA GEMMs
// C[M,N] = A[M,K] @ B[K,N] + bias ; BT stored [N,K] bf16.
// 8 waves/block, each wave owns a 16x64 tile. The 64x32 B tile (4KB) is
// shared by all waves -> staged once per K-step into LDS with async
// global->LDS copies (double-buffered), eliminating 8x redundant B traffic.
// Cooperative copy: 256 threads x 16B = 4KB. Last K-step peeled so the
// steady-state loop has an unconditional prefetch (no branch in hot loop).
// Requires K % 32 == 0 and K >= 64 (true for all call sites: K = 1024).
__global__ __launch_bounds__(256) void mha_gemm_bf16A(const __bf16* __restrict__ A,
                                                      const __bf16* __restrict__ BT,
                                                      const float* __restrict__ bias,
                                                      float* __restrict__ C,
                                                      int M, int N, int K) {
    __shared__ __bf16 ldsB[2][64 * 32];
    int t    = threadIdx.x;
    int lane = t & 31;
    int wave = t >> 5;
    int m0 = (blockIdx.x * 8 + wave) * 16;
    int n0 = blockIdx.y * 64;
    int nl = t >> 2;            // 0..63  (n-local row of B tile)
    int kq = (t & 3) * 8;       // 0,8,16,24 (k-local, 8 bf16 = 16B)
    const __bf16* bsrc = BT + (size_t)(n0 + nl) * K + kq;

    async_ld_lds_b128(&ldsB[0][nl * 32 + kq], bsrc);
    wait_asynccnt0();
    __syncthreads();

    v8f acc[4] = {};
    int buf = 0;
    for (int k0 = 0; k0 < K - 32; k0 += 32) {
        async_ld_lds_b128(&ldsB[buf ^ 1][nl * 32 + kq], bsrc + k0 + 32);
        v16bf a = load_a_bf16(A, K, m0, k0, lane);
#pragma unroll
        for (int j = 0; j < 4; ++j) {
            v16bf b = load_b_lds(&ldsB[buf][0], j * 16, lane);
            acc[j] = WMMA_BF16(a, b, acc[j]);
        }
        wait_asynccnt0();       // own async copy into buf^1 landed
        __syncthreads();        // all waves landed + done reading buf
        buf ^= 1;
    }
    {   // peeled final K-step: no prefetch, no trailing barrier
        v16bf a = load_a_bf16(A, K, m0, K - 32, lane);
#pragma unroll
        for (int j = 0; j < 4; ++j) {
            v16bf b = load_b_lds(&ldsB[buf][0], j * 16, lane);
            acc[j] = WMMA_BF16(a, b, acc[j]);
        }
    }
    int mrow = m0 + (lane >> 4) * 8;
#pragma unroll
    for (int j = 0; j < 4; ++j) {
        int n = n0 + j * 16 + (lane & 15);
        float bv = bias ? bias[n] : 0.f;
#pragma unroll
        for (int r = 0; r < 8; ++r)
            C[(size_t)(mrow + r) * N + n] = acc[j][r] + bv;
    }
}

__global__ __launch_bounds__(256) void mha_gemm_f32A(const float* __restrict__ A,
                                                     const __bf16* __restrict__ BT,
                                                     const float* __restrict__ bias,
                                                     float* __restrict__ C,
                                                     int M, int N, int K) {
    __shared__ __bf16 ldsB[2][64 * 32];
    int t    = threadIdx.x;
    int lane = t & 31;
    int wave = t >> 5;
    int m0 = (blockIdx.x * 8 + wave) * 16;
    int n0 = blockIdx.y * 64;
    int nl = t >> 2;
    int kq = (t & 3) * 8;
    const __bf16* bsrc = BT + (size_t)(n0 + nl) * K + kq;

    async_ld_lds_b128(&ldsB[0][nl * 32 + kq], bsrc);
    wait_asynccnt0();
    __syncthreads();

    v8f acc[4] = {};
    int buf = 0;
    for (int k0 = 0; k0 < K - 32; k0 += 32) {
        async_ld_lds_b128(&ldsB[buf ^ 1][nl * 32 + kq], bsrc + k0 + 32);
        v16bf a = load_a_f32(A, K, m0, k0, lane);
#pragma unroll
        for (int j = 0; j < 4; ++j) {
            v16bf b = load_b_lds(&ldsB[buf][0], j * 16, lane);
            acc[j] = WMMA_BF16(a, b, acc[j]);
        }
        wait_asynccnt0();
        __syncthreads();
        buf ^= 1;
    }
    {   // peeled final K-step
        v16bf a = load_a_f32(A, K, m0, K - 32, lane);
#pragma unroll
        for (int j = 0; j < 4; ++j) {
            v16bf b = load_b_lds(&ldsB[buf][0], j * 16, lane);
            acc[j] = WMMA_BF16(a, b, acc[j]);
        }
    }
    int mrow = m0 + (lane >> 4) * 8;
#pragma unroll
    for (int j = 0; j < 4; ++j) {
        int n = n0 + j * 16 + (lane & 15);
        float bv = bias ? bias[n] : 0.f;
#pragma unroll
        for (int r = 0; r < 8; ++r)
            C[(size_t)(mrow + r) * N + n] = acc[j][r] + bv;
    }
}

// ----------------------------------------------------- RoPE + scale + V^T
// qp/kp/vp: fp32 [B,S,H*DH]; out qh/kh: bf16 [B,H,S,DH]; vT: bf16 [B,H,DH,S]
__global__ void mha_rope_kernel(const float* __restrict__ qp, const float* __restrict__ kp,
                                const float* __restrict__ vp,
                                const float* __restrict__ cosT, const float* __restrict__ sinT,
                                __bf16* __restrict__ qh, __bf16* __restrict__ kh,
                                __bf16* __restrict__ vT) {
    int idx = blockIdx.x * blockDim.x + threadIdx.x;   // B*S*H*(DH/2) = 1<<20
    if (idx >= cB * cS * cH * (cDH / 2)) return;
    int i = idx & 31;
    int h = (idx >> 5) & 15;
    int s = (idx >> 9) & 1023;
    int b = idx >> 19;
    float c  = cosT[s * 32 + i];
    float sn = sinT[s * 32 + i];
    size_t prow = ((size_t)b * cS + s) * cD + h * cDH;
    float qe = qp[prow + 2 * i], qo = qp[prow + 2 * i + 1];
    float ke = kp[prow + 2 * i], ko = kp[prow + 2 * i + 1];
    float qre = (qe * c - qo * sn) * 0.125f;   // 1/sqrt(64)
    float qim = (qe * sn + qo * c) * 0.125f;
    float kre =  ke * c - ko * sn;
    float kim =  ke * sn + ko * c;
    size_t hrow = (((size_t)b * cH + h) * cS + s) * cDH;
    qh[hrow + 2 * i]     = (__bf16)qre;
    qh[hrow + 2 * i + 1] = (__bf16)qim;
    kh[hrow + 2 * i]     = (__bf16)kre;
    kh[hrow + 2 * i + 1] = (__bf16)kim;
    size_t vbase = ((size_t)b * cH + h) * (size_t)cDH * cS;
    vT[vbase + (size_t)(2 * i)     * cS + s] = (__bf16)vp[prow + 2 * i];
    vT[vbase + (size_t)(2 * i + 1) * cS + s] = (__bf16)vp[prow + 2 * i + 1];
}

// ------------------------------------------- qrel[row,t] = q_row . rel_emb[t]
__global__ __launch_bounds__(128) void mha_qrel_kernel(const __bf16* __restrict__ qh,
                                                       const float* __restrict__ rel_emb,
                                                       float* __restrict__ qrel) {
    __shared__ float qv[cDH];
    size_t row = blockIdx.x;                       // bh*S + s
    int tid = threadIdx.x;
    if (tid < cDH) qv[tid] = (float)qh[row * cDH + tid];
    __syncthreads();
    if (tid < cT) {
        const float* re = rel_emb + (size_t)tid * cDH;
        float acc = 0.f;
#pragma unroll 8
        for (int d = 0; d < cDH; ++d) acc += qv[d] * re[d];
        qrel[row * cT + tid] = acc;
    }
}

// ----------------------------- scores[b,h,q,k] = q.k (WMMA) + qrel[bucket]
__global__ __launch_bounds__(256) void mha_scores_kernel(const __bf16* __restrict__ qh,
                                                         const __bf16* __restrict__ kh,
                                                         const float* __restrict__ qrel,
                                                         float* __restrict__ scores) {
    int lane = threadIdx.x & 31;
    int wave = threadIdx.x >> 5;
    int bh = blockIdx.z;
    int m0 = (blockIdx.x * 8 + wave) * 16;
    int n0 = blockIdx.y * 64;
    const __bf16* A  = qh + (size_t)bh * cS * cDH;
    const __bf16* BT = kh + (size_t)bh * cS * cDH;
    v8f acc[4] = {};
#pragma unroll
    for (int k0 = 0; k0 < cDH; k0 += 32) {
        v16bf a = load_a_bf16(A, cDH, m0, k0, lane);
#pragma unroll
        for (int j = 0; j < 4; ++j) {
            v16bf b = load_b_bf16T(BT, cDH, n0 + j * 16, k0, lane);
            acc[j] = WMMA_BF16(a, b, acc[j]);
        }
    }
    int mrow = m0 + (lane >> 4) * 8;
#pragma unroll
    for (int j = 0; j < 4; ++j) {
        int n = n0 + j * 16 + (lane & 15);
#pragma unroll
        for (int r = 0; r < 8; ++r) {
            int m = mrow + r;
            int d = n - m;
            d = max(-32, min(32, d));
            float s = acc[j][r] + qrel[((size_t)bh * cS + m) * cT + d + 32];
            scores[((size_t)bh * cS + m) * cS + n] = s;
        }
    }
}

// ---------------------------------------------- entmax 1.5 over each row
// One 1024-thread block per row: max-reduce, bitonic descending sort,
// Hillis-Steele cumsum (x, x^2), tau*, y = clip(x - tau*)^2, bucket sums.
__global__ __launch_bounds__(1024) void mha_entmax_kernel(float* __restrict__ scores,
                                                          float* __restrict__ bucket) {
    __shared__ float xs[cS];
    __shared__ float c1[cS];
    __shared__ float c2[cS];
    __shared__ float bkt[80];
    int tid = threadIdx.x;
    size_t row = blockIdx.x;                // bh*S + q
    int q = (int)(row & (cS - 1));
    float* srow = scores + row * cS;
    float x = srow[tid];

    c1[tid] = x; __syncthreads();
    for (int s = 512; s > 0; s >>= 1) {
        if (tid < s) c1[tid] = fmaxf(c1[tid], c1[tid + s]);
        __syncthreads();
    }
    float mx = c1[0]; __syncthreads();
    float xv = (x - mx) * 0.5f;
    xs[tid] = xv; __syncthreads();

    for (int ksz = 2; ksz <= cS; ksz <<= 1)
        for (int j = ksz >> 1; j > 0; j >>= 1) {
            int ixj = tid ^ j;
            if (ixj > tid) {
                float a = xs[tid], b = xs[ixj];
                bool descBlk = ((tid & ksz) == 0);
                if (descBlk ? (a < b) : (a > b)) { xs[tid] = b; xs[ixj] = a; }
            }
            __syncthreads();
        }

    float sx = xs[tid];
    c1[tid] = sx; c2[tid] = sx * sx; __syncthreads();
    for (int off = 1; off < cS; off <<= 1) {
        float t1 = 0.f, t2 = 0.f;
        if (tid >= off) { t1 = c1[tid - off]; t2 = c2[tid - off]; }
        __syncthreads();
        c1[tid] += t1; c2[tid] += t2;
        __syncthreads();
    }
    float rho   = (float)(tid + 1);
    float mean  = c1[tid] / rho;
    float msq   = c2[tid] / rho;
    float ss    = rho * (msq - mean * mean);
    float delta = (1.0f - ss) / rho;
    float tau   = mean - sqrtf(fmaxf(delta, 0.f));
    __syncthreads();
    c1[tid] = (tau <= sx) ? 1.f : 0.f;
    xs[tid] = tau;
    __syncthreads();
    for (int s = 512; s > 0; s >>= 1) {
        if (tid < s) c1[tid] += c1[tid + s];
        __syncthreads();
    }
    int support = (int)(c1[0] + 0.5f);
    float tau_star = xs[support - 1];
    float y = fmaxf(xv - tau_star, 0.f);
    y *= y;
    srow[tid] = y;                           // attn overwrites scores in place

    if (tid < 80) bkt[tid] = 0.f;
    __syncthreads();
    int d = tid - q;
    d = max(-32, min(32, d));
    atomicAdd(&bkt[d + 32], y);
    __syncthreads();
    if (tid < cT) bucket[row * cT + tid] = bkt[tid];
}

// ----------- ctx[b,h,q,:] = attn @ v (WMMA) + bucket[q,:] @ rel_emb
__global__ __launch_bounds__(256) void mha_ctx_kernel(const float* __restrict__ attn,
                                                      const __bf16* __restrict__ vT,
                                                      const float* __restrict__ bucket,
                                                      const float* __restrict__ rel_emb,
                                                      float* __restrict__ ctx) {
    int lane = threadIdx.x & 31;
    int wave = threadIdx.x >> 5;
    int bh = blockIdx.y;
    int b = bh >> 4, h = bh & 15;
    int m0 = (blockIdx.x * 8 + wave) * 16;
    const float*  A  = attn + (size_t)bh * cS * cS;
    const __bf16* BT = vT   + (size_t)bh * cDH * cS;   // [DH, S]
    v8f acc[4] = {};
    for (int k0 = 0; k0 < cS; k0 += 32) {
        v16bf a = load_a_f32(A, cS, m0, k0, lane);
#pragma unroll
        for (int j = 0; j < 4; ++j) {                  // 4*16 = DH=64 columns
            v16bf bfr = load_b_bf16T(BT, cS, j * 16, k0, lane);
            acc[j] = WMMA_BF16(a, bfr, acc[j]);
        }
    }
    int mrow = m0 + (lane >> 4) * 8;
    int nlo = lane & 15;
    for (int t = 0; t < cT; ++t) {
        float re[4];
#pragma unroll
        for (int j = 0; j < 4; ++j) re[j] = rel_emb[(size_t)t * cDH + j * 16 + nlo];
#pragma unroll
        for (int r = 0; r < 8; ++r) {
            float bkv = bucket[((size_t)bh * cS + mrow + r) * cT + t];
#pragma unroll
            for (int j = 0; j < 4; ++j) acc[j][r] += bkv * re[j];
        }
    }
#pragma unroll
    for (int j = 0; j < 4; ++j) {
        int n = j * 16 + nlo;
#pragma unroll
        for (int r = 0; r < 8; ++r) {
            int m = mrow + r;
            ctx[((size_t)b * cS + m) * cD + h * cDH + n] = acc[j][r];
        }
    }
}

// ---------------------------------------------------------------- launcher
extern "C" void kernel_launch(void* const* d_in, const int* in_sizes, int n_in,
                              void* d_out, int out_size, void* d_ws, size_t ws_size,
                              hipStream_t stream) {
    (void)in_sizes; (void)n_in; (void)out_size; (void)ws_size;
    const float* x       = (const float*)d_in[0];
    const float* w_q     = (const float*)d_in[1];
    const float* b_q     = (const float*)d_in[2];
    const float* w_k     = (const float*)d_in[3];
    const float* b_k     = (const float*)d_in[4];
    const float* w_v     = (const float*)d_in[5];
    const float* b_v     = (const float*)d_in[6];
    const float* w_o     = (const float*)d_in[7];
    const float* b_o     = (const float*)d_in[8];
    const float* rel_emb = (const float*)d_in[9];
    const float* rcos    = (const float*)d_in[10];
    const float* rsin    = (const float*)d_in[11];
    float* out = (float*)d_out;

    char* ws = (char*)d_ws;
    size_t off = 0;
    auto carve = [&](size_t bytes) -> char* {
        char* p = ws + off;
        off += (bytes + 255) & ~(size_t)255;
        return p;
    };
    __bf16* xb    = (__bf16*)carve((size_t)cBS * cD * 2);
    __bf16* wqT   = (__bf16*)carve((size_t)cD * cD * 2);
    __bf16* wkT   = (__bf16*)carve((size_t)cD * cD * 2);
    __bf16* wvT   = (__bf16*)carve((size_t)cD * cD * 2);
    __bf16* woT   = (__bf16*)carve((size_t)cD * cD * 2);
    float*  qp    = (float*) carve((size_t)cBS * cD * 4);
    float*  kp    = (float*) carve((size_t)cBS * cD * 4);
    float*  vp    = (float*) carve((size_t)cBS * cD * 4);
    __bf16* qh    = (__bf16*)carve((size_t)cBH * cS * cDH * 2);
    __bf16* kh    = (__bf16*)carve((size_t)cBH * cS * cDH * 2);
    __bf16* vT    = (__bf16*)carve((size_t)cBH * cDH * cS * 2);
    float*  qrel  = (float*) carve((size_t)cBH * cS * cT * 4);
    float*  sc    = (float*) carve((size_t)cBH * cS * cS * 4);   // scores/attn
    float*  bkt   = (float*) carve((size_t)cBH * cS * cT * 4);
    float*  ctx   = (float*) carve((size_t)cBS * cD * 4);

    // 1) casts / weight transposes
    mha_castx_kernel<<<(cBS * cD + 255) / 256, 256, 0, stream>>>(x, xb, cBS * cD);
    dim3 tg(cD / 32, cD / 32);
    mha_castT_kernel<<<tg, 256, 0, stream>>>(w_q, wqT);
    mha_castT_kernel<<<tg, 256, 0, stream>>>(w_k, wkT);
    mha_castT_kernel<<<tg, 256, 0, stream>>>(w_v, wvT);
    mha_castT_kernel<<<tg, 256, 0, stream>>>(w_o, woT);

    // 2) Q/K/V projections (WMMA bf16, async-LDS-staged B, fp32 accumulate)
    dim3 gg(cBS / 128, cD / 64);        // (16,16): 8 waves x 16 rows, 64 cols
    mha_gemm_bf16A<<<gg, 256, 0, stream>>>(xb, wqT, b_q, qp, cBS, cD, cD);
    mha_gemm_bf16A<<<gg, 256, 0, stream>>>(xb, wkT, b_k, kp, cBS, cD, cD);
    mha_gemm_bf16A<<<gg, 256, 0, stream>>>(xb, wvT, b_v, vp, cBS, cD, cD);

    // 3) RoPE + q-scale + head transpose (+ V^T for WMMA B operand)
    mha_rope_kernel<<<(cB * cS * cH * (cDH / 2)) / 256, 256, 0, stream>>>(
        qp, kp, vp, rcos, rsin, qh, kh, vT);

    // 4) relative-position score buckets
    mha_qrel_kernel<<<cBH * cS, 128, 0, stream>>>(qh, rel_emb, qrel);

    // 5) scores = q k^T + rel (WMMA)
    mha_scores_kernel<<<dim3(cS / 128, cS / 64, cBH), 256, 0, stream>>>(qh, kh, qrel, sc);

    // 6) entmax-1.5 row-wise (sort + scan in LDS), attn overwrites scores
    mha_entmax_kernel<<<cBH * cS, 1024, 0, stream>>>(sc, bkt);

    // 7) ctx = attn @ v + bucket @ rel_emb (WMMA)
    mha_ctx_kernel<<<dim3(cS / 128, cBH), 256, 0, stream>>>(sc, vT, bkt, rel_emb, ctx);

    // 8) output projection (WMMA, fp32 A converted on load)
    mha_gemm_f32A<<<gg, 256, 0, stream>>>(ctx, woT, b_o, out, cBS, cD, cD);
}
